// PureTriXButterfly_63806034149896
// MI455X (gfx1250) — matrix operand order repaired and checked by reading, not compiled, not run on hardware
//
#include <hip/hip_runtime.h>
#include <math.h>
#include <stdint.h>

// ---------------------------------------------------------------------------
// Problem constants (from reference)
// ---------------------------------------------------------------------------
#define DV   16        // value range -> 256 distinct (a,b) pairs
#define NR   256       // LUT rows = DV*DV
#define DM   1024      // d_model
#define NT   16        // tiles
#define NLAY 3
#define NTOP 4
#define BATCH 16384
#define NOUT 11

// LDS tile geometry: K-step 32, padded row stride 40 bf16 (80 B):
// 16B-aligned for b128 ds reads, 20-dword bank stride -> conflict-free.
#define KSTEP 32
#define LPAD  40

// 1 = use gfx1250 GLOBAL_LOAD_ASYNC_TO_LDS_B128 (ASYNCcnt) for the bf16
// activation tile copy; 0 = plain load+ds_store fallback.
#define USE_ASYNC_LDS 1

typedef __attribute__((ext_vector_type(16))) __bf16 v16bf;
typedef __attribute__((ext_vector_type(8)))  float  v8f;

// ---------------------------------------------------------------------------
// Helpers
// ---------------------------------------------------------------------------
__device__ __forceinline__ float gelu_f(float x) {
  return 0.5f * x * (1.0f + erff(x * 0.70710678118654752440f));
}

__device__ __forceinline__ float block_sum256(float v, float* red, int tid) {
  red[tid] = v;
  __syncthreads();
  #pragma unroll
  for (int s = 128; s > 0; s >>= 1) {
    if (tid < s) red[tid] += red[tid + s];
    __syncthreads();
  }
  float r = red[0];
  __syncthreads();
  return r;
}

// A-fragment (16x32 bf16) from an LDS tile laid out [row][LPAD] row-major.
// ISA 7.12.2: lanes 0-15 row M=lane, K={0..7,16..23}; lanes 16-31 row M=lane-16,
// K={8..15,24..31}. Pairs contiguous -> dword reads, 16B-aligned groups.
__device__ __forceinline__ v16bf frag_a(const __bf16 (*AL)[LPAD], int m0, int lane) {
  int m  = m0 + (lane & 15);
  int kh = (lane & 16) ? 4 : 0;
  const uint32_t* p = (const uint32_t*)&AL[m][0];
  union { v16bf v; uint32_t u[8]; } a;
  #pragma unroll
  for (int j = 0; j < 4; ++j) a.u[j]     = p[kh + j];
  #pragma unroll
  for (int j = 0; j < 4; ++j) a.u[4 + j] = p[kh + 8 + j];
  return a.v;
}

// B-fragment (32x16 bf16) from LDS tile stored TRANSPOSED as [n][LPAD] (k
// contiguous). lanes 0-15: col N=lane, K={0..7,16..23}; lanes 16-31: col N=lane-16,
// K={8..15,24..31} (same K order as frag_a).
__device__ __forceinline__ v16bf frag_b(const __bf16 (*WS)[LPAD], int nl, int lane) {
  int n  = nl + (lane & 15);
  int kh = (lane & 16) ? 4 : 0;
  const uint32_t* p = (const uint32_t*)&WS[n][0];
  union { v16bf v; uint32_t u[8]; } b;
  #pragma unroll
  for (int j = 0; j < 4; ++j) b.u[j]     = p[kh + j];
  #pragma unroll
  for (int j = 0; j < 4; ++j) b.u[4 + j] = p[kh + 8 + j];
  return b.v;
}

// Async (or fallback) copy of one 64-byte activation row slice into LDS.
__device__ __forceinline__ void copy_row_async(const __bf16* gbase, uint32_t goff_bytes,
                                               __bf16* lrow) {
#if USE_ASYNC_LDS
  uint32_t lds = (uint32_t)(uintptr_t)lrow;  // low 32 bits of flat LDS addr = LDS offset
  asm volatile("global_load_async_to_lds_b128 %0, %1, %2"
               :: "v"(lds),       "v"(goff_bytes),       "s"(gbase) : "memory");
  asm volatile("global_load_async_to_lds_b128 %0, %1, %2"
               :: "v"(lds + 16u), "v"(goff_bytes + 16u), "s"(gbase) : "memory");
  asm volatile("global_load_async_to_lds_b128 %0, %1, %2"
               :: "v"(lds + 32u), "v"(goff_bytes + 32u), "s"(gbase) : "memory");
  asm volatile("global_load_async_to_lds_b128 %0, %1, %2"
               :: "v"(lds + 48u), "v"(goff_bytes + 48u), "s"(gbase) : "memory");
#else
  const uint4* src = (const uint4*)((const char*)gbase + goff_bytes);
  uint4* dst = (uint4*)lrow;
  #pragma unroll
  for (int c = 0; c < 4; ++c) dst[c] = src[c];
#endif
}

__device__ __forceinline__ void wait_async0() {
#if USE_ASYNC_LDS
  asm volatile("s_wait_asynccnt 0x0" ::: "memory");
#endif
}

// Cooperative W block (32K x 64N fp32) load into registers (coalesced float4).
struct WRegs { float4 a, b; };
__device__ __forceinline__ WRegs load_w(const float* Wt, int kk, int ncol0, int tid) {
  WRegs w;
  int i0 = tid, i1 = tid + 256;                       // 512 float4 slots
  w.a = *(const float4*)(Wt + (size_t)(kk + (i0 >> 4)) * DM + ncol0 + (i0 & 15) * 4);
  w.b = *(const float4*)(Wt + (size_t)(kk + (i1 >> 4)) * DM + ncol0 + (i1 & 15) * 4);
  return w;
}
// Convert + store transposed into LDS [n][LPAD].
__device__ __forceinline__ void store_w(const WRegs& w, __bf16 (*WS)[LPAD], int tid) {
  int i0 = tid, i1 = tid + 256;
  int k0 = i0 >> 4, q0 = (i0 & 15) * 4;
  int k1 = i1 >> 4, q1 = (i1 & 15) * 4;
  WS[q0 + 0][k0] = (__bf16)w.a.x;  WS[q0 + 1][k0] = (__bf16)w.a.y;
  WS[q0 + 2][k0] = (__bf16)w.a.z;  WS[q0 + 3][k0] = (__bf16)w.a.w;
  WS[q1 + 0][k1] = (__bf16)w.b.x;  WS[q1 + 1][k1] = (__bf16)w.b.y;
  WS[q1 + 2][k1] = (__bf16)w.b.z;  WS[q1 + 3][k1] = (__bf16)w.b.w;
}

// ---------------------------------------------------------------------------
// 1) Encode: fourier(a,b) -> @Win + bias -> LN -> GELU, for all 256 LUT rows
// ---------------------------------------------------------------------------
__global__ void k_encode(const float* __restrict__ Win, const float* __restrict__ bin_,
                         const float* __restrict__ g_in, const float* __restrict__ b_ln,
                         float* __restrict__ Xf, __bf16* __restrict__ Xh) {
  __shared__ float feat[32];
  __shared__ float red[256];
  int r = blockIdx.x, tid = threadIdx.x;
  if (tid < 32) {
    int src = (tid < 16) ? (r & 15) : (r >> 4);
    int k   = tid & 7;
    float ang = (float)src * (6.28318530717958647692f / 16.0f) * (float)(1 << k);
    feat[tid] = (tid & 8) ? cosf(ang) : sinf(ang);
  }
  __syncthreads();
  float v[4];
  #pragma unroll
  for (int q = 0; q < 4; ++q) {
    int d = tid + q * 256;
    float acc = bin_[d];
    #pragma unroll
    for (int j = 0; j < 32; ++j) acc += feat[j] * Win[j * DM + d];
    v[q] = acc;
  }
  float mean = block_sum256(v[0] + v[1] + v[2] + v[3], red, tid) * (1.0f / DM);
  float q2 = 0.f;
  #pragma unroll
  for (int q = 0; q < 4; ++q) { float dl = v[q] - mean; q2 += dl * dl; }
  float var = block_sum256(q2, red, tid) * (1.0f / DM);
  float inv = rsqrtf(var + 1e-5f);
  #pragma unroll
  for (int q = 0; q < 4; ++q) {
    int d = tid + q * 256;
    float xn = (v[q] - mean) * inv * g_in[d] + b_ln[d];
    float ge = gelu_f(xn);
    Xf[(size_t)r * DM + d] = ge;
    Xh[(size_t)r * DM + d] = (__bf16)ge;
  }
}

// ---------------------------------------------------------------------------
// 2) Router (parallel): one block per LUT row; 256-thread partial reduction.
// ---------------------------------------------------------------------------
__global__ void k_router(const float* __restrict__ Xf, const float* __restrict__ Wr_l,
                         float* __restrict__ gates) {
  __shared__ float part[256][NT + 1];   // +1 pad
  int r = blockIdx.x, tid = threadIdx.x;
  float lg[NT];
  #pragma unroll
  for (int t = 0; t < NT; ++t) lg[t] = 0.f;
  const float* xr = Xf + (size_t)r * DM;
  #pragma unroll
  for (int q = 0; q < 4; ++q) {
    int d = tid + q * 256;
    float xv = xr[d];
    #pragma unroll
    for (int t = 0; t < NT; ++t) lg[t] += xv * Wr_l[d * NT + t];
  }
  #pragma unroll
  for (int t = 0; t < NT; ++t) part[tid][t] = lg[t];
  __syncthreads();
  for (int s = 128; s > 0; s >>= 1) {
    if (tid < s) {
      #pragma unroll
      for (int t = 0; t < NT; ++t) part[tid][t] += part[tid + s][t];
    }
    __syncthreads();
  }
  if (tid == 0) {
    float p[NT];
    float m = part[0][0];
    #pragma unroll
    for (int t = 0; t < NT; ++t) { p[t] = part[0][t]; m = fmaxf(m, p[t]); }
    float s = 0.f;
    #pragma unroll
    for (int t = 0; t < NT; ++t) { p[t] = expf(p[t] - m); s += p[t]; }
    float invs = 1.0f / s;
    #pragma unroll
    for (int t = 0; t < NT; ++t) p[t] *= invs;
    float g[NT]; bool used[NT];
    #pragma unroll
    for (int t = 0; t < NT; ++t) { g[t] = 0.f; used[t] = false; }
    for (int k = 0; k < NTOP; ++k) {      // first-index wins on ties, like top_k
      float best = -1.f; int bi = 0;
      #pragma unroll
      for (int t = 0; t < NT; ++t)
        if (!used[t] && p[t] > best) { best = p[t]; bi = t; }
      used[bi] = true; g[bi] = best;
    }
    #pragma unroll
    for (int t = 0; t < NT; ++t) gates[r * NT + t] = g[t];
  }
}

// ---------------------------------------------------------------------------
// 3) FFN stage 1: Hg[t] = gate[:,t] * gelu(X @ W1[t] + b1[t])   (bf16 out)
//    grid = (16 N-blocks, 16 tiles), 256 threads = 8 waves.
//    Double-buffered LDS: A tile via async global->LDS copy, W tile via
//    coalesced fp32 loads + cvt + transposed store.
// ---------------------------------------------------------------------------
__global__ void k_ffn1(const __bf16* __restrict__ Xh, const float* __restrict__ W1_l,
                       const float* __restrict__ b1_l, const float* __restrict__ gates,
                       __bf16* __restrict__ Hg) {
  __shared__ __bf16 AL[2][NR][LPAD];   // 40 KB
  __shared__ __bf16 WS[2][64][LPAD];   // 10 KB
  const int t    = blockIdx.y;
  const int nb   = blockIdx.x;
  const int tid  = threadIdx.x;
  const int lane = tid & 31;
  const int w    = tid >> 5;
  const int nt   = w & 3;
  const int mh   = w >> 2;
  const int ncol0 = nb * 64;
  const float* Wt = W1_l + (size_t)t * DM * DM;

  const v8f vzero = {0.f, 0.f, 0.f, 0.f, 0.f, 0.f, 0.f, 0.f};
  v8f acc[8];
  #pragma unroll
  for (int mt = 0; mt < 8; ++mt) acc[mt] = vzero;

  // prologue: stage step 0
  WRegs wr = load_w(Wt, 0, ncol0, tid);
  copy_row_async(Xh, (uint32_t)((tid * DM + 0) * 2), &AL[0][tid][0]);

  const int NSTEP = DM / KSTEP;  // 32
  int buf = 0;
  for (int s = 0; s < NSTEP; ++s) {
    wait_async0();                 // AL[buf] copy complete (this wave's issues)
    store_w(wr, WS[buf], tid);     // WS[buf]: cvt + transposed store
    __syncthreads();               // publish AL[buf], WS[buf]; retire buf^1 readers
    if (s + 1 < NSTEP) {           // stage step s+1 into buf^1 (overlaps compute)
      int kk = (s + 1) * KSTEP;
      copy_row_async(Xh, (uint32_t)((tid * DM + kk) * 2), &AL[buf ^ 1][tid][0]);
      wr = load_w(Wt, kk, ncol0, tid);
      __builtin_prefetch(Wt + (size_t)(kk + KSTEP) * DM + ncol0 + (tid & 15) * 4, 0, 1);
    }
    v16bf bf = frag_b(WS[buf], nt * 16, lane);
    #pragma unroll
    for (int mt = 0; mt < 8; ++mt) {
      int m0 = (mh * 8 + mt) * 16;
      v16bf af = frag_a(AL[buf], m0, lane);
      acc[mt] = __builtin_amdgcn_wmma_f32_16x16x32_bf16(
          false, af, false, bf, (short)0, acc[mt], false, false);
    }
    buf ^= 1;
  }

  const int col = ncol0 + nt * 16 + (lane & 15);
  const int mo  = (lane & 16) ? 8 : 0;
  const float bias = b1_l[t * DM + col];
  #pragma unroll
  for (int mt = 0; mt < 8; ++mt) {
    int m0 = (mh * 8 + mt) * 16;
    #pragma unroll
    for (int j = 0; j < 8; ++j) {
      int row = m0 + mo + j;
      float vv = gelu_f(acc[mt][j] + bias) * gates[row * NT + t];
      Hg[((size_t)t * NR + row) * DM + col] = (__bf16)vv;
    }
  }
}

// ---------------------------------------------------------------------------
// 4) FFN stage 2: P[kc] = Hg_cat(256 x 16384) @ W2_cat(16384 x 1024), split-K.
//    grid = (16 N-blocks, 8 K-chunks). Deterministic partial buffers.
// ---------------------------------------------------------------------------
__global__ void k_ffn2(const __bf16* __restrict__ Hg, const float* __restrict__ W2_l,
                       float* __restrict__ P) {
  __shared__ __bf16 AL[2][NR][LPAD];
  __shared__ __bf16 WS[2][64][LPAD];
  const int nb   = blockIdx.x;
  const int kc   = blockIdx.y;
  const int tid  = threadIdx.x;
  const int lane = tid & 31;
  const int w    = tid >> 5;
  const int nt   = w & 3;
  const int mh   = w >> 2;
  const int ncol0 = nb * 64;
  const int k0    = kc * 2048;

  const v8f vzero = {0.f, 0.f, 0.f, 0.f, 0.f, 0.f, 0.f, 0.f};
  v8f acc[8];
  #pragma unroll
  for (int mt = 0; mt < 8; ++mt) acc[mt] = vzero;

  // A global offset for K-step kk: tile t=kk/1024, f=kk%1024; row = tid.
  WRegs wr = load_w(W2_l, k0, ncol0, tid);
  {
    int tt = k0 >> 10, ff = k0 & (DM - 1);
    copy_row_async(Hg, (uint32_t)((((tt * NR + tid) * DM) + ff) * 2), &AL[0][tid][0]);
  }

  const int NSTEP = 2048 / KSTEP;  // 64
  int buf = 0;
  for (int s = 0; s < NSTEP; ++s) {
    wait_async0();
    store_w(wr, WS[buf], tid);
    __syncthreads();
    if (s + 1 < NSTEP) {
      int kk = k0 + (s + 1) * KSTEP;
      int tt = kk >> 10, ff = kk & (DM - 1);
      copy_row_async(Hg, (uint32_t)((((tt * NR + tid) * DM) + ff) * 2), &AL[buf ^ 1][tid][0]);
      wr = load_w(W2_l, kk, ncol0, tid);
      __builtin_prefetch(W2_l + (size_t)(kk + KSTEP) * DM + ncol0 + (tid & 15) * 4, 0, 1);
    }
    v16bf bf = frag_b(WS[buf], nt * 16, lane);
    #pragma unroll
    for (int mt = 0; mt < 8; ++mt) {
      int m0 = (mh * 8 + mt) * 16;
      v16bf af = frag_a(AL[buf], m0, lane);
      acc[mt] = __builtin_amdgcn_wmma_f32_16x16x32_bf16(
          false, af, false, bf, (short)0, acc[mt], false, false);
    }
    buf ^= 1;
  }

  const int col = ncol0 + nt * 16 + (lane & 15);
  const int mo  = (lane & 16) ? 8 : 0;
  float* Pk = P + (size_t)kc * NR * DM;
  #pragma unroll
  for (int mt = 0; mt < 8; ++mt) {
    int m0 = (mh * 8 + mt) * 16;
    #pragma unroll
    for (int j = 0; j < 8; ++j)
      Pk[(size_t)(m0 + mo + j) * DM + col] = acc[mt][j];
  }
}

// ---------------------------------------------------------------------------
// 5) Finish: x = LN(x + sum_kc P[kc] + sum_t g[t]*b2[t]) * g_l + b_l
// ---------------------------------------------------------------------------
__global__ void k_finish(const float* __restrict__ P, const float* __restrict__ gates,
                         const float* __restrict__ b2_l, const float* __restrict__ g_l,
                         const float* __restrict__ b_l,
                         float* __restrict__ Xf, __bf16* __restrict__ Xh) {
  __shared__ float red[256];
  __shared__ float gsh[NT];
  int r = blockIdx.x, tid = threadIdx.x;
  if (tid < NT) gsh[tid] = gates[r * NT + tid];
  __syncthreads();
  float v[4];
  #pragma unroll
  for (int q = 0; q < 4; ++q) {
    int d = tid + q * 256;
    float s = Xf[(size_t)r * DM + d];
    #pragma unroll
    for (int kc = 0; kc < 8; ++kc)          // fixed-order split-K reduction
      s += P[((size_t)kc * NR + r) * DM + d];
    #pragma unroll
    for (int t = 0; t < NT; ++t) s += gsh[t] * b2_l[t * DM + d];
    v[q] = s;
  }
  float mean = block_sum256(v[0] + v[1] + v[2] + v[3], red, tid) * (1.0f / DM);
  float q2 = 0.f;
  #pragma unroll
  for (int q = 0; q < 4; ++q) { float dl = v[q] - mean; q2 += dl * dl; }
  float var = block_sum256(q2, red, tid) * (1.0f / DM);
  float inv = rsqrtf(var + 1e-5f);
  #pragma unroll
  for (int q = 0; q < 4; ++q) {
    int d = tid + q * 256;
    float xn = (v[q] - mean) * inv * g_l[d] + b_l[d];
    Xf[(size_t)r * DM + d] = xn;
    Xh[(size_t)r * DM + d] = (__bf16)xn;
  }
}

// ---------------------------------------------------------------------------
// 6) Heads: lut[r] = [gelu(x@Ws1+bs1)@Ws2+bs2 , gelu(x@Wd1+bd1)@Wd2+bd2]
// ---------------------------------------------------------------------------
__global__ void k_heads(const float* __restrict__ Xf,
                        const float* __restrict__ Ws1, const float* __restrict__ bs1,
                        const float* __restrict__ Ws2, const float* __restrict__ bs2,
                        const float* __restrict__ Wd1, const float* __restrict__ bd1,
                        const float* __restrict__ Wd2, const float* __restrict__ bd2,
                        float* __restrict__ lut) {
  __shared__ float hs[512];
  __shared__ float hd[512];
  int r = blockIdx.x, tid = threadIdx.x;
  const float* x = Xf + (size_t)r * DM;
  #pragma unroll
  for (int rep = 0; rep < 2; ++rep) {
    int j = tid + rep * 256;
    float as = bs1[j], ad = bd1[j];
    for (int d = 0; d < DM; ++d) {
      float xv = x[d];
      as += xv * Ws1[d * 512 + j];
      ad += xv * Wd1[d * 512 + j];
    }
    hs[j] = gelu_f(as);
    hd[j] = gelu_f(ad);
  }
  __syncthreads();
  if (tid < NOUT) {
    float acc;
    if (tid < 5) {
      acc = bs2[tid];
      for (int k = 0; k < 512; ++k) acc += hs[k] * Ws2[k * 5 + tid];
    } else {
      int j = tid - 5;
      acc = bd2[j];
      for (int k = 0; k < 512; ++k) acc += hd[k] * Wd2[k * 6 + j];
    }
    lut[r * NOUT + tid] = acc;
  }
}

// ---------------------------------------------------------------------------
// 7) Scatter: out[b] = lut[a[b] + 16*b[b]]
// ---------------------------------------------------------------------------
__global__ void k_scatter(const int* __restrict__ a, const int* __restrict__ bb,
                          const float* __restrict__ lut, float* __restrict__ out,
                          int total) {
  int idx = blockIdx.x * blockDim.x + threadIdx.x;
  if (idx >= total) return;
  int row = idx / NOUT, j = idx - row * NOUT;
  int r = (a[row] & 15) | ((bb[row] & 15) << 4);
  out[idx] = lut[r * NOUT + j];
}

// ---------------------------------------------------------------------------
// Launch
// ---------------------------------------------------------------------------
extern "C" void kernel_launch(void* const* d_in, const int* in_sizes, int n_in,
                              void* d_out, int out_size, void* d_ws, size_t ws_size,
                              hipStream_t stream) {
  (void)in_sizes; (void)n_in; (void)out_size; (void)ws_size;
  const int*   a     = (const int*)  d_in[0];
  const int*   b     = (const int*)  d_in[1];
  const float* Win   = (const float*)d_in[2];
  const float* bin_  = (const float*)d_in[3];
  const float* g_in  = (const float*)d_in[4];
  const float* b_lni = (const float*)d_in[5];
  const float* Wr    = (const float*)d_in[6];
  const float* W1    = (const float*)d_in[7];
  const float* b1    = (const float*)d_in[8];
  const float* W2    = (const float*)d_in[9];
  const float* b2    = (const float*)d_in[10];
  const float* g_l   = (const float*)d_in[11];
  const float* b_l   = (const float*)d_in[12];
  const float* Ws1   = (const float*)d_in[13];
  const float* bs1   = (const float*)d_in[14];
  const float* Ws2   = (const float*)d_in[15];
  const float* bs2   = (const float*)d_in[16];
  const float* Wd1   = (const float*)d_in[17];
  const float* bd1   = (const float*)d_in[18];
  const float* Wd2   = (const float*)d_in[19];
  const float* bd2   = (const float*)d_in[20];

  char* ws = (char*)d_ws;
  float*  Xf    = (float*) (ws + 0);                         // 1 MB
  __bf16* Xh    = (__bf16*)(ws + (1u << 20));                // 0.5 MB
  float*  gates = (float*) (ws + (1u << 20) + (512u << 10)); // 16 KB
  float*  lut   = (float*) (ws + (1u << 20) + (528u << 10)); // 11 KB
  __bf16* Hg    = (__bf16*)(ws + (2u << 20));                // 8 MB
  float*  P     = (float*) (ws + (10u << 20));               // 8 MB

  k_encode<<<NR, 256, 0, stream>>>(Win, bin_, g_in, b_lni, Xf, Xh);

  for (int i = 0; i < NLAY; ++i) {
    const float* Wr_l = Wr + (size_t)i * DM * NT;
    const float* W1_l = W1 + (size_t)i * NT * DM * DM;
    const float* b1_l = b1 + (size_t)i * NT * DM;
    const float* W2_l = W2 + (size_t)i * NT * DM * DM;
    const float* b2_l = b2 + (size_t)i * NT * DM;

    k_router<<<NR, 256, 0, stream>>>(Xf, Wr_l, gates);
    k_ffn1<<<dim3(16, 16), 256, 0, stream>>>(Xh, W1_l, b1_l, gates, Hg);
    k_ffn2<<<dim3(16, 8), 256, 0, stream>>>(Hg, W2_l, P);
    k_finish<<<NR, 256, 0, stream>>>(P, gates, b2_l, g_l + i * DM, b_l + i * DM,
                                     Xf, Xh);
  }

  k_heads<<<NR, 256, 0, stream>>>(Xf, Ws1, bs1, Ws2, bs2, Wd1, bd1, Wd2, bd2, lut);

  int total = BATCH * NOUT;
  k_scatter<<<(total + 255) / 256, 256, 0, stream>>>(a, b, lut, (float*)d_out, total);
}
